// MambaBlock_63058709840310
// MI455X (gfx1250) — matrix operand order repaired and checked by reading, not compiled, not run on hardware
//
#include <hip/hip_runtime.h>
#include <hip/hip_bf16.h>

// ---- Mamba block forward for MI455X (gfx1250, wave32, WMMA bf16) ----
// GEMMs: v_wmma_f32_16x16x32_bf16 with 4x4 (or 4x2) register tiling per wave
// -> 1 global_load_b128 per WMMA, keeping the matrix pipe (not VMEM) the limiter.
// Fragment addressing follows CDNA5 ISA 7.12.2 so fragments are contiguous
// 16B runs of row-major memory.

typedef unsigned short bf16_t;
typedef __attribute__((ext_vector_type(16))) __bf16 v16bf;
typedef __attribute__((ext_vector_type(8)))  float  v8f;

__device__ __forceinline__ bf16_t f32_to_bf16(float f) {
  unsigned u = __float_as_uint(f);
  unsigned r = 0x7FFFu + ((u >> 16) & 1u);   // round-to-nearest-even
  return (bf16_t)((u + r) >> 16);
}

// ---------------- f32 -> bf16 conversion (vectorized, n % 4 == 0) ----------------
__global__ __launch_bounds__(256)
void cvt_f32_to_bf16_kernel(const float* __restrict__ in,
                            unsigned int* __restrict__ out, long long n4) {
  long long i = (long long)blockIdx.x * 256 + threadIdx.x;
  if (i < n4) {
    float4 v = ((const float4*)in)[i];
    unsigned int lo = (unsigned)f32_to_bf16(v.x) | ((unsigned)f32_to_bf16(v.y) << 16);
    unsigned int hi = (unsigned)f32_to_bf16(v.z) | ((unsigned)f32_to_bf16(v.w) << 16);
    out[i * 2 + 0] = lo;
    out[i * 2 + 1] = hi;
  }
}

// ---------------- WMMA GEMM: out[T,N] = A[T,K](bf16) * W[N,K]^T(bf16) ----------------
// Wave computes MB x NB grid of 16x16 tiles. EPI==1: softplus(x+bias[n]).
union Frag { uint4 q[2]; v16bf v; };

template <int MB, int NB, int EPI>
__global__ __launch_bounds__(256)
void wmma_gemm_kernel(const bf16_t* __restrict__ A, int lda,
                      const bf16_t* __restrict__ W, int ldw,
                      float* __restrict__ out, int ldo,
                      const float* __restrict__ bias,
                      int T, int N, int K) {
  const int lane = threadIdx.x & 31;
  const int half = lane >> 4;            // 0: lanes 0-15, 1: lanes 16-31
  const int l15  = lane & 15;
  const long long wtile = (long long)blockIdx.x * 8 + (threadIdx.x >> 5);
  const int nWaveN = (N >> 4) / NB;
  const long long nWTiles = (long long)((T >> 4) / MB) * nWaveN;
  if (wtile >= nWTiles) return;          // wave-uniform: EXEC stays all-ones
  const int mt0 = (int)(wtile / nWaveN) * MB;
  const int nt0 = (int)(wtile % nWaveN) * NB;

  const bf16_t* arow[MB];
  const bf16_t* brow[NB];
#pragma unroll
  for (int i = 0; i < MB; ++i)
    arow[i] = A + (size_t)((mt0 + i) * 16 + l15) * lda;
#pragma unroll
  for (int j = 0; j < NB; ++j)
    brow[j] = W + (size_t)((nt0 + j) * 16 + l15) * ldw;

  const int aoff = half * 8;             // A: K chunks [aoff,aoff+8) and [16+aoff,..)
  const int boff = half * 8;             // B: K chunks [boff,boff+8) and [16+boff,..)

  v8f acc[MB][NB];
#pragma unroll
  for (int i = 0; i < MB; ++i)
#pragma unroll
    for (int j = 0; j < NB; ++j) acc[i][j] = (v8f){};

  for (int k0 = 0; k0 < K; k0 += 32) {
    Frag af[MB], bfr[NB];
#pragma unroll
    for (int i = 0; i < MB; ++i) {
      af[i].q[0] = *(const uint4*)(arow[i] + k0 + aoff);
      af[i].q[1] = *(const uint4*)(arow[i] + k0 + 16 + aoff);
    }
#pragma unroll
    for (int j = 0; j < NB; ++j) {
      bfr[j].q[0] = *(const uint4*)(brow[j] + k0 + boff);
      bfr[j].q[1] = *(const uint4*)(brow[j] + k0 + 16 + boff);
    }
    if (k0 + 288 <= K) {                 // gfx1250 global_prefetch_b8, near-cache
#pragma unroll
      for (int i = 0; i < MB; ++i) __builtin_prefetch(arow[i] + k0 + 256, 0, 3);
#pragma unroll
      for (int j = 0; j < NB; ++j) __builtin_prefetch(brow[j] + k0 + 256, 0, 3);
    }
#pragma unroll
    for (int i = 0; i < MB; ++i)
#pragma unroll
      for (int j = 0; j < NB; ++j)
        acc[i][j] = __builtin_amdgcn_wmma_f32_16x16x32_bf16(
            false, af[i].v, false, bfr[j].v, (short)0, acc[i][j], false, false);
  }

  // C/D layout: element r -> M = r + 8*half, N = lane&15
#pragma unroll
  for (int i = 0; i < MB; ++i) {
#pragma unroll
    for (int j = 0; j < NB; ++j) {
      const int ncol = (nt0 + j) * 16 + l15;
      float bia = (EPI == 1) ? bias[ncol] : 0.0f;
      const size_t base = (size_t)((mt0 + i) * 16 + half * 8) * ldo + ncol;
#pragma unroll
      for (int r = 0; r < 8; ++r) {
        float v = acc[i][j][r];
        if (EPI == 1) {                  // softplus(v + bias)
          float x = v + bia;
          v = (x > 20.0f) ? x : log1pf(__expf(x));
        }
        out[base + (size_t)r * ldo] = v;
      }
    }
  }
}

// ---------------- depthwise causal conv1d (K=4) + SiLU ----------------
__global__ __launch_bounds__(256)
void conv_silu_kernel(const float* __restrict__ xar,
                      const float* __restrict__ cw, const float* __restrict__ cb,
                      float* __restrict__ u, bf16_t* __restrict__ ub) {
  long long i = (long long)blockIdx.x * 256 + threadIdx.x;  // over 4096*4096
  int d = (int)(i & 4095);
  long long t = i >> 12;                                    // b*2048 + l
  int l = (int)(t & 2047);
  float acc = cb[d];
  const float* w = cw + d * 4;
#pragma unroll
  for (int j = 0; j < 4; ++j) {
    int ll = l - 3 + j;
    if (ll >= 0) acc += xar[(t - 3 + j) * 8192 + d] * w[j];
  }
  float s = acc * (1.0f / (1.0f + __expf(-acc)));           // SiLU
  u[i] = s;
  ub[i] = f32_to_bf16(s);
}

// ---------------- selective scan + D skip + SiLU gate ----------------
// One thread per (b,d); h[16] in registers; serial over L=2048.
// B/C rows are read by all lanes at the same address -> hardware broadcast;
// cheaper than per-step LDS staging (which would need 2 barriers x 2048 steps).
__global__ __launch_bounds__(256)
void scan_kernel(const float* __restrict__ delta, const float* __restrict__ u,
                 const float* __restrict__ xdbl,  const float* __restrict__ xar,
                 const float* __restrict__ A_log, const float* __restrict__ Dp,
                 bf16_t* __restrict__ ygb) {
  const int d = (blockIdx.x & 15) * 256 + threadIdx.x;      // 16 blocks per batch
  const int b = blockIdx.x >> 4;
  float a[16], h[16];
#pragma unroll
  for (int n = 0; n < 16; ++n) { a[n] = -__expf(A_log[d * 16 + n]); h[n] = 0.0f; }
  const float Dd = Dp[d];

  for (int l = 0; l < 2048; ++l) {
    const size_t row = (size_t)b * 2048 + l;
    float dlt = delta[row * 4096 + d];
    float uu  = u[row * 4096 + d];
    float Bv[16], Cv[16];
    const float4* Bp = (const float4*)(xdbl + row * 160 + 128);
    const float4* Cp = (const float4*)(xdbl + row * 160 + 144);
#pragma unroll
    for (int q = 0; q < 4; ++q) { *(float4*)(Bv + 4 * q) = Bp[q]; *(float4*)(Cv + 4 * q) = Cp[q]; }

    float du = dlt * uu;
    float y = 0.0f;
#pragma unroll
    for (int n = 0; n < 16; ++n) {
      float dA = __expf(dlt * a[n]);
      h[n] = dA * h[n] + du * Bv[n];
      y += h[n] * Cv[n];
    }
    y += uu * Dd;                                           // D skip
    float r = xar[row * 8192 + 4096 + d];                   // res (gate input)
    y *= r * (1.0f / (1.0f + __expf(-r)));                  // * silu(res)
    ygb[row * 4096 + d] = f32_to_bf16(y);

    // prefetch next timestep's streams (gfx1250 global_prefetch_b8)
    __builtin_prefetch(delta + (row + 1) * 4096 + d, 0, 3);
    __builtin_prefetch(u     + (row + 1) * 4096 + d, 0, 3);
    __builtin_prefetch(xar   + (row + 1) * 8192 + 4096 + d, 0, 3);
  }
}

// ---------------- host launcher ----------------
extern "C" void kernel_launch(void* const* d_in, const int* in_sizes, int n_in,
                              void* d_out, int out_size, void* d_ws, size_t ws_size,
                              hipStream_t stream) {
  (void)in_sizes; (void)n_in; (void)out_size; (void)ws_size;
  const float* x    = (const float*)d_in[0];   // [2,2048,2048]
  const float* wi   = (const float*)d_in[1];   // [8192,2048]
  const float* cw   = (const float*)d_in[2];   // [4096,1,4]
  const float* cb   = (const float*)d_in[3];   // [4096]
  const float* wx   = (const float*)d_in[4];   // [160,4096]
  const float* wd   = (const float*)d_in[5];   // [4096,128]
  const float* dbias= (const float*)d_in[6];   // [4096]
  const float* Alog = (const float*)d_in[7];   // [4096,16]
  const float* Dp   = (const float*)d_in[8];   // [4096]
  const float* wo   = (const float*)d_in[9];   // [2048,4096]

  const int T = 4096;                          // B*L tokens
  char* p = (char*)d_ws;
  auto alloc = [&](size_t bytes) -> void* {
    void* r = (void*)p; p += (bytes + 255) & ~(size_t)255; return r;
  };
  bf16_t* xb   = (bf16_t*)alloc((size_t)T * 2048 * 2);
  bf16_t* wib  = (bf16_t*)alloc((size_t)8192 * 2048 * 2);
  bf16_t* wxb  = (bf16_t*)alloc((size_t)160 * 4096 * 2);
  bf16_t* wdb  = (bf16_t*)alloc((size_t)4096 * 128 * 2);
  bf16_t* wob  = (bf16_t*)alloc((size_t)2048 * 4096 * 2);
  float*  xar  = (float*) alloc((size_t)T * 8192 * 4);   // in_proj out [xi | res]
  float*  u    = (float*) alloc((size_t)T * 4096 * 4);
  bf16_t* ub   = (bf16_t*)alloc((size_t)T * 4096 * 2);
  float*  xdbl = (float*) alloc((size_t)T * 160 * 4);    // [dlt | B | C]
  bf16_t* xdb  = (bf16_t*)alloc((size_t)T * 160 * 2);
  float*  delta= (float*) alloc((size_t)T * 4096 * 4);
  bf16_t* ygb  = (bf16_t*)alloc((size_t)T * 4096 * 2);

  auto cvt = [&](const float* src, bf16_t* dst, long long n) {
    long long n4 = n / 4;
    cvt_f32_to_bf16_kernel<<<(unsigned)((n4 + 255) / 256), 256, 0, stream>>>(
        src, (unsigned int*)dst, n4);
  };
  cvt(x,  xb,  (long long)T * 2048);
  cvt(wi, wib, (long long)8192 * 2048);
  cvt(wx, wxb, (long long)160 * 4096);
  cvt(wd, wdb, (long long)4096 * 128);
  cvt(wo, wob, (long long)2048 * 4096);

  // in_proj: [T,2048] x [8192,2048]^T -> xar [T,8192]   (64 x 128 wave tiles)
  {
    long long wt = (long long)(T / 64) * (8192 / 64);
    wmma_gemm_kernel<4, 4, 0><<<(unsigned)((wt + 7) / 8), 256, 0, stream>>>(
        xb, 2048, wib, 2048, xar, 8192, nullptr, T, 8192, 2048);
  }
  // depthwise conv + SiLU -> u, ub
  conv_silu_kernel<<<(unsigned)((long long)T * 4096 / 256), 256, 0, stream>>>(
      xar, cw, cb, u, ub);
  // x_proj: [T,4096] x [160,4096]^T -> xdbl [T,160]     (64 x 5 wave tiles, NB=2)
  {
    long long wt = (long long)(T / 64) * (160 / 32);
    wmma_gemm_kernel<4, 2, 0><<<(unsigned)((wt + 7) / 8), 256, 0, stream>>>(
        ub, 4096, wxb, 4096, xdbl, 160, nullptr, T, 160, 4096);
  }
  cvt(xdbl, xdb, (long long)T * 160);
  // dt_proj + softplus: [T,128](lda=160) x [4096,128]^T -> delta [T,4096]
  {
    long long wt = (long long)(T / 64) * (4096 / 64);
    wmma_gemm_kernel<4, 4, 1><<<(unsigned)((wt + 7) / 8), 256, 0, stream>>>(
        xdb, 160, wdb, 128, delta, 4096, dbias, T, 4096, 128);
  }
  // selective scan + skip + gate -> ygb (bf16)
  scan_kernel<<<32, 256, 0, stream>>>(delta, u, xdbl, xar, Alog, Dp, ygb);
  // out_proj: [T,4096] x [2048,4096]^T -> d_out [T,2048]
  {
    long long wt = (long long)(T / 64) * (2048 / 64);
    wmma_gemm_kernel<4, 4, 0><<<(unsigned)((wt + 7) / 8), 256, 0, stream>>>(
        ygb, 4096, wob, 4096, (float*)d_out, 2048, nullptr, T, 4096, 4096);
  }
}
